// EnhancedVectorQuantizer_66829691126164
// MI455X (gfx1250) — compile-verified, matrix operand order checked by reading, not run on hardware
//
#include <hip/hip_runtime.h>
#include <hip/hip_bf16.h>
#include <math.h>

// ---------------- problem constants ----------------
#define N_ROWS   65536      // 16*4096
#define DIM      256
#define KCODES   2048
#define D4       (DIM/4)    // 64 float4 per row
#define TM_WG    128        // input rows per workgroup (8 waves x 16 rows)
#define ESTAGE   128        // codes staged in LDS per stage
#define NSTAGES  (KCODES/ESTAGE)   // 16
#define Q_ELEMS  (N_ROWS*DIM)      // 16777216
#define EPS_F    1e-8f

typedef __attribute__((ext_vector_type(16))) __bf16 v16bf;
typedef __attribute__((ext_vector_type(8)))  float  v8f;
typedef __attribute__((ext_vector_type(4)))  int    i4v;

// round-to-nearest-even f32 -> bf16, packed pair into one u32
__device__ __forceinline__ unsigned int bf16pk(float a, float b) {
    unsigned int ua = __float_as_uint(a);
    ua = (ua + 0x7FFFu + ((ua >> 16) & 1u)) >> 16;
    unsigned int ub = __float_as_uint(b);
    ub = (ub + 0x7FFFu + ((ub >> 16) & 1u)) >> 16;
    return (ua & 0xFFFFu) | (ub << 16);
}

// Load a 16x32 bf16 WMMA A/B fragment (ISA 7.12.2 layout) from a row-major
// [row][256] bf16 array. Caller passes pointer at (row*256 + dk*32 + half*8):
// VGPR0..3 hold k = 8*half + 0..7, VGPR4..7 hold k = 8*half + 16..23.
__device__ __forceinline__ v16bf load_frag(const unsigned short* p) {
    union { v16bf v; uint4 u[2]; } f;
    f.u[0] = *(const uint4*)(p);
    f.u[1] = *(const uint4*)(p + 16);
    return f.v;
}

__device__ __forceinline__ v8f zero_v8f() {
    v8f c;
#pragma unroll
    for (int i = 0; i < 8; ++i) c[i] = 0.0f;
    return c;
}

// async copy one 128-codes-x-256-dims bf16 stage (64 KB) into an LDS buffer:
// 256 threads x 16 chunks of 16B, consecutive lanes touch consecutive 16B.
__device__ __forceinline__ void issue_stage(const unsigned short* embB, int es,
                                            unsigned short* buf, int tid) {
    const unsigned short* g = embB + (size_t)es * ESTAGE * DIM;
#pragma unroll
    for (int i = 0; i < 16; ++i) {
        const int chunk = i * 256 + tid;            // 0..4095
        __builtin_amdgcn_global_load_async_to_lds_b128(
            (__attribute__((address_space(1))) i4v*)(g + chunk * 8),
            (__attribute__((address_space(3))) i4v*)(buf + chunk * 8),
            0, 0);
    }
}

// ---------------- kernel 0: zero scratch accumulators ----------------
__global__ void vq_zero_kernel(float* counts, float* scal) {
    for (int k = threadIdx.x; k < KCODES; k += 256) counts[k] = 0.0f;
    if (threadIdx.x < 16) scal[threadIdx.x] = 0.0f;
}

// ---------------- kernel 1: emb row norms + bf16 copies ----------------
// grid 256 x 256thr: one wave per emb row. Emits:
//   eNorm2[k] = ||e_k||^2, scal[0] += ||e_k||,
//   embB    = bf16(emb)          (for async-fed distance GEMM)
//   normedB = bf16(emb/||e_k||)  (for orth-reg GEMM)
__global__ __launch_bounds__(256) void vq_norm_kernel(
    const float* __restrict__ emb, float* __restrict__ eNorm2,
    unsigned short* __restrict__ embB, unsigned short* __restrict__ normedB,
    float* __restrict__ scal) {
    const int lane = threadIdx.x & 31;
    const int wave = threadIdx.x >> 5;
    const int row  = blockIdx.x * 8 + wave;

    const float4* e4 = (const float4*)(emb + (size_t)row * DIM);
    float4 v0 = e4[lane * 2 + 0];
    float4 v1 = e4[lane * 2 + 1];
    float s = v0.x*v0.x + v0.y*v0.y + v0.z*v0.z + v0.w*v0.w
            + v1.x*v1.x + v1.y*v1.y + v1.z*v1.z + v1.w*v1.w;
#pragma unroll
    for (int m = 16; m >= 1; m >>= 1) s += __shfl_xor(s, m);
    s = __shfl(s, 0);
    const float rn  = sqrtf(s);
    const float inv = 1.0f / fmaxf(rn, 1e-12f);
    if (lane == 0) {
        eNorm2[row] = s;
        atomicAdd(&scal[0], rn);   // sum of row norms -> l2_reg
    }
    uint4 raw;
    raw.x = bf16pk(v0.x, v0.y);
    raw.y = bf16pk(v0.z, v0.w);
    raw.z = bf16pk(v1.x, v1.y);
    raw.w = bf16pk(v1.z, v1.w);
    *(uint4*)&embB[(size_t)row * DIM + lane * 8] = raw;
    uint4 pk;
    pk.x = bf16pk(v0.x*inv, v0.y*inv);
    pk.y = bf16pk(v0.z*inv, v0.w*inv);
    pk.z = bf16pk(v1.x*inv, v1.y*inv);
    pk.w = bf16pk(v1.z*inv, v1.w*inv);
    *(uint4*)&normedB[(size_t)row * DIM + lane * 8] = pk;
}

// ---------------- kernel 2: fused distance GEMM + argmin ----------------
// grid 512 x 256thr; each WG: 128 rows vs all 2048 codes.
// Codebook (bf16) streams through a double-buffered 2x64KB LDS window via
// GLOBAL_LOAD_ASYNC_TO_LDS_B128; one stage always in flight behind the WMMAs.
__global__ __launch_bounds__(256) void vq_dist_argmin_kernel(
    const float* __restrict__ x, const unsigned short* __restrict__ embB,
    const float* __restrict__ eNorm2, int* __restrict__ idxOut) {
    __shared__ unsigned short smE[2][ESTAGE * DIM];   // 2 x 64 KB

    const int tid  = threadIdx.x;
    const int lane = tid & 31;
    const int wave = tid >> 5;
    const int half = lane >> 4;
    const int ln   = lane & 15;
    const int rowBase = blockIdx.x * TM_WG;

    // kick off codebook stage 0 while we stage x and build A fragments
    issue_stage(embB, 0, smE[0], tid);

    // ---- stage x tile (128 rows) into smE[1] as bf16 ----
    {
        const float4* x4 = (const float4*)(x + (size_t)rowBase * DIM);
#pragma unroll
        for (int i = 0; i < 32; ++i) {
            const int e4i = tid + i * 256;          // 0..8191
            float4 v = x4[e4i];
            uint2 pk;
            pk.x = bf16pk(v.x, v.y);
            pk.y = bf16pk(v.z, v.w);
            *(uint2*)&smE[1][e4i * 4] = pk;
        }
    }
    __syncthreads();

    // ---- hoist this wave's A fragments (16 rows x 256 dims) into VGPRs ----
    v16bf afr[8];
    {
        const int arow = wave * 16 + ln;
#pragma unroll
        for (int dk = 0; dk < 8; ++dk)
            afr[dk] = load_frag(&smE[1][arow * DIM + dk * 32 + half * 8]);
    }
    __syncthreads();                 // everyone done reading smE[1]
    issue_stage(embB, 1, smE[1], tid);

    float minv[8];
    int   mini[8];
#pragma unroll
    for (int r = 0; r < 8; ++r) { minv[r] = 3.4e38f; mini[r] = 0; }

    // ---- sweep codebook stages, double-buffered ----
    for (int es = 0; es < NSTAGES; ++es) {
        if (es < NSTAGES - 1) __builtin_amdgcn_s_wait_asynccnt(16);  // stage es landed
        else                  __builtin_amdgcn_s_wait_asynccnt(0);
        __syncthreads();
        const unsigned short* buf = smE[es & 1];

        for (int ct = 0; ct < 8; ++ct) {
            v8f c = zero_v8f();
            const int brow = ct * 16 + ln;
#pragma unroll
            for (int dk = 0; dk < 8; ++dk) {
                v16bf bfr = load_frag(&buf[brow * DIM + dk * 32 + half * 8]);
                c = __builtin_amdgcn_wmma_f32_16x16x32_bf16(
                        false, afr[dk], false, bfr, (short)0, c, false, false);
            }
            const int   code = es * ESTAGE + ct * 16 + ln;
            const float en   = eNorm2[code];
#pragma unroll
            for (int r = 0; r < 8; ++r) {
                const float dist = en - 2.0f * c[r];   // ||x||^2 constant per row
                if (dist < minv[r]) { minv[r] = dist; mini[r] = code; }
            }
        }
        __syncthreads();             // all waves done reading smE[es&1]
        if (es + 2 < NSTAGES) issue_stage(embB, es + 2, smE[es & 1], tid);
    }

    // ---- argmin across the 16 lanes holding the same row ----
#pragma unroll
    for (int r = 0; r < 8; ++r) {
        float v  = minv[r];
        int   ix = mini[r];
#pragma unroll
        for (int m = 8; m >= 1; m >>= 1) {
            const float ov = __shfl_xor(v, m);
            const int   oi = __shfl_xor(ix, m);
            if (ov < v || (ov == v && oi < ix)) { v = ov; ix = oi; }
        }
        if (ln == 0)
            idxOut[rowBase + wave * 16 + r + half * 8] = ix;
    }
}

// ---------------- kernel 3: gather + SSE + counts + idx-as-float ----------------
// grid 16384 x 256thr: one float4 per thread over N*D (fp32 gather for output)
__global__ __launch_bounds__(256) void vq_gather_kernel(
    const float* __restrict__ x, const float* __restrict__ emb,
    const int* __restrict__ idx, float* __restrict__ outQ,
    float* __restrict__ outIdxF, float* __restrict__ counts,
    float* __restrict__ scal) {
    const int g  = blockIdx.x * 256 + threadIdx.x;  // 0 .. N*64-1
    const int n  = g >> 6;
    const int c4 = g & 63;
    const int k  = idx[n];

    float4 e  = ((const float4*)emb)[(size_t)k * D4 + c4];
    float4 xv = ((const float4*)x)[g];
    ((float4*)outQ)[g] = e;   // quantized_st forward value == quantized

    const float dx = e.x - xv.x, dy = e.y - xv.y, dz = e.z - xv.z, dw = e.w - xv.w;
    float s = dx*dx + dy*dy + dz*dz + dw*dw;

    if (c4 == 0) {
        atomicAdd(&counts[k], 1.0f);
        outIdxF[n] = (float)k;
    }

    const int lane = threadIdx.x & 31;
    const int wave = threadIdx.x >> 5;
#pragma unroll
    for (int m = 16; m >= 1; m >>= 1) s += __shfl_xor(s, m);
    __shared__ float red[8];
    if (lane == 0) red[wave] = s;
    __syncthreads();
    if (threadIdx.x == 0) {
        float t = 0.0f;
#pragma unroll
        for (int w = 0; w < 8; ++w) t += red[w];
        atomicAdd(&scal[1], t);   // sum of squared diffs
    }
}

// ---------------- kernel 4: orth reg via WMMA on normed bf16 ----------------
// grid 2048 x 256thr: one 16x16 sim tile per wave (128x128 tiles total)
__global__ __launch_bounds__(256) void vq_orth_kernel(
    const unsigned short* __restrict__ nb, float* __restrict__ scal) {
    const int lane = threadIdx.x & 31;
    const int wave = threadIdx.x >> 5;
    const int half = lane >> 4;
    const int ln   = lane & 15;
    const int t    = blockIdx.x * 8 + wave;   // 0..16383
    const int ti   = t >> 7;                  // row tile
    const int tj   = t & 127;                 // col tile

    v8f c = zero_v8f();
    const int arow = ti * 16 + ln;
    const int brow = tj * 16 + ln;
#pragma unroll
    for (int dk = 0; dk < 8; ++dk) {
        v16bf a = load_frag(&nb[(size_t)arow * DIM + dk * 32 + half * 8]);
        v16bf b = load_frag(&nb[(size_t)brow * DIM + dk * 32 + half * 8]);
        c = __builtin_amdgcn_wmma_f32_16x16x32_bf16(
                false, a, false, b, (short)0, c, false, false);
    }
    float s = 0.0f;
#pragma unroll
    for (int r = 0; r < 8; ++r) {
        const int row = ti * 16 + r + half * 8;
        const int col = tj * 16 + ln;
        if (row != col) s += fabsf(c[r]);
    }
#pragma unroll
    for (int m = 16; m >= 1; m >>= 1) s += __shfl_xor(s, m);
    if (lane == 0) atomicAdd(&scal[2], s);
}

// ---------------- kernel 5: finalize scalars ----------------
__global__ __launch_bounds__(256) void vq_finalize_kernel(
    const float* __restrict__ counts, const float* __restrict__ scal,
    float* __restrict__ outScalars /* [0]=loss, [1]=perplexity */) {
    __shared__ float r1[8], r2[8];
    __shared__ float sTotal;
    const int tid  = threadIdx.x;
    const int lane = tid & 31;
    const int wave = tid >> 5;

    float t = 0.0f;
    for (int k = tid; k < KCODES; k += 256) t += counts[k];
#pragma unroll
    for (int m = 16; m >= 1; m >>= 1) t += __shfl_xor(t, m);
    if (lane == 0) r1[wave] = t;
    __syncthreads();
    if (tid == 0) {
        float tt = 0.0f;
        for (int w = 0; w < 8; ++w) tt += r1[w];
        sTotal = tt;
    }
    __syncthreads();
    const float total = sTotal;
    const float tuni  = 1.0f / (float)KCODES;

    float kl = 0.0f, pl = 0.0f;
    for (int k = tid; k < KCODES; k += 256) {
        const float p = counts[k] / total;
        kl += (p + EPS_F) * logf((p + EPS_F) / (tuni + EPS_F));
        if (p > 0.0f) pl += p * logf(p);
    }
#pragma unroll
    for (int m = 16; m >= 1; m >>= 1) { kl += __shfl_xor(kl, m); pl += __shfl_xor(pl, m); }
    __syncthreads();
    if (lane == 0) { r1[wave] = kl; r2[wave] = pl; }
    __syncthreads();
    if (tid == 0) {
        float klt = 0.0f, plt = 0.0f;
        for (int w = 0; w < 8; ++w) { klt += r1[w]; plt += r2[w]; }
        klt = fminf(klt, 100.0f);
        const float l2_reg   = fminf(scal[0] / (float)KCODES, 10.0f);
        const float orth_reg = fminf(scal[2] / ((float)KCODES * (float)KCODES), 10.0f);
        const float mse      = scal[1] / ((float)N_ROWS * (float)DIM);
        const float vq_loss  = (1.0f + 0.25f) * mse;   // q_latent + 0.25*e_latent (equal fwd)
        const float loss = fminf(vq_loss + 0.1f * klt + 0.01f * (l2_reg + orth_reg), 100.0f);
        outScalars[0] = loss;
        outScalars[1] = expf(-plt);
    }
}

// ---------------- launch ----------------
extern "C" void kernel_launch(void* const* d_in, const int* in_sizes, int n_in,
                              void* d_out, int out_size, void* d_ws, size_t ws_size,
                              hipStream_t stream) {
    const float* x   = (const float*)d_in[0];   // [16,4096,256]
    const float* emb = (const float*)d_in[1];   // [2048,256]
    float* out = (float*)d_out;
    float* outQ       = out;                    // 16777216
    float* outScalars = out + Q_ELEMS;          // loss, perplexity
    float* outIdxF    = out + Q_ELEMS + 2;      // 65536

    // workspace layout (~2.4 MB)
    float* ws      = (float*)d_ws;
    float* eNorm2  = ws;                          // 2048 f32
    float* counts  = ws + KCODES;                 // 2048 f32
    float* scal    = ws + 2 * KCODES;             // [0]=sum||e||, [1]=sse, [2]=sum|off|
    unsigned short* embB    = (unsigned short*)(ws + 2 * KCODES + 16);  // 2048*256 bf16
    unsigned short* normedB = embB + (size_t)KCODES * DIM;              // 2048*256 bf16
    int* idxI = (int*)(normedB + (size_t)KCODES * DIM);                 // 65536 i32

    vq_zero_kernel<<<1, 256, 0, stream>>>(counts, scal);
    vq_norm_kernel<<<KCODES / 8, 256, 0, stream>>>(emb, eNorm2, embB, normedB, scal);
    vq_dist_argmin_kernel<<<N_ROWS / TM_WG, 256, 0, stream>>>(x, embB, eNorm2, idxI);
    vq_gather_kernel<<<(N_ROWS * D4) / 256, 256, 0, stream>>>(x, emb, idxI, outQ,
                                                              outIdxF, counts, scal);
    vq_orth_kernel<<<(KCODES / 16) * (KCODES / 16) / 8, 256, 0, stream>>>(normedB, scal);
    vq_finalize_kernel<<<1, 256, 0, stream>>>(counts, scal, outScalars);
}
